// DotAttention_74251394614054
// MI455X (gfx1250) — matrix operand order repaired and checked by reading, not compiled, not run on hardware
//
#include <hip/hip_runtime.h>
#include <hip/hip_bf16.h>

// Problem constants (match reference)
#define BB  32
#define LQ  2048
#define LK  2048
#define DD  1024

typedef __attribute__((ext_vector_type(16))) __bf16 v16bf;
typedef __attribute__((ext_vector_type(8)))  float  v8f;

union Frag16 { v16bf v; uint4 u[2]; };
union U4U16  { uint4 q; unsigned short s[8]; };

// f32 -> bf16 round-to-nearest-even, scalar (epilogue / softmax use)
__device__ __forceinline__ unsigned short f2bfu(float f) {
    union { float f; unsigned int u; } c; c.f = f;
    unsigned int r = c.u + 0x7FFFu + ((c.u >> 16) & 1u);
    return (unsigned short)(r >> 16);
}
// RNE-round a f32 bit pattern so bf16 result sits in bits [31:16]
__device__ __forceinline__ unsigned int rne_hi(float f) {
    union { float f; unsigned int u; } c; c.f = f;
    return c.u + 0x7FFFu + ((c.u >> 16) & 1u);
}
// pack two RNE-rounded bf16 with one v_perm_b32
__device__ __forceinline__ unsigned int f2bf_pk(float a, float b) {
    return __builtin_amdgcn_perm(rne_hi(b), rne_hi(a), 0x07060302u);
}

// CDNA5 async global->LDS copy, 16B per lane, tracked by ASYNCcnt.
// GVS addressing: mem = SADDR(64b sgpr) + VADDR(32b vgpr) ; LDS dest = VDST vgpr.
__device__ __forceinline__ void async_b128(unsigned lds_addr, unsigned goff,
                                           unsigned long long base) {
    asm volatile("global_load_async_to_lds_b128 %0, %1, %2"
                 :: "v"(lds_addr), "v"(goff), "s"(base) : "memory");
}
__device__ __forceinline__ void wait_async0() {
    asm volatile("s_wait_asynccnt 0x0" ::: "memory");
}
__device__ __forceinline__ unsigned lds_off(const void* p) {
    return (unsigned)(uintptr_t)p;   // generic addr low 32 bits == LDS byte offset
}

// ---------------------------------------------------------------------------
// Streaming f32 -> bf16 convert (8 elements per thread, fully coalesced)
// ---------------------------------------------------------------------------
__global__ __launch_bounds__(256)
void cvt_f32_bf16_kernel(const float* __restrict__ src, unsigned short* __restrict__ dst)
{
    const size_t idx = (size_t)blockIdx.x * 256 + threadIdx.x;
    const float4 a = ((const float4*)src)[2 * idx];
    const float4 b = ((const float4*)src)[2 * idx + 1];
    uint4 o;
    o.x = f2bf_pk(a.x, a.y);
    o.y = f2bf_pk(a.z, a.w);
    o.z = f2bf_pk(b.x, b.y);
    o.w = f2bf_pk(b.z, b.w);
    ((uint4*)dst)[idx] = o;
}

// ---------------------------------------------------------------------------
// Tiled bf16 WMMA GEMM:  C[b] = A[b] * op(B[b])
//   A: [M x Kd] row-major (lda), bf16 -> async-DMA'd to LDS (no staging regs).
//   B_KMAJOR = true : B is [Kd x N] row-major (ldb) -> reg-staged transpose
//   B_KMAJOR = false: B is [N x Kd] row-major (ldb) -> async-DMA'd (C = A*B^T)
//   OUT_BF16 selects bf16(ushort) vs f32 output.
// Tiles: BM=256, BN=128, BK=32; 256 threads = 8 waves (4x2), wave tile 64x64
// (4x4 WMMA accumulators). LDS double-buffered; ASYNCcnt-paced pipeline.
// ---------------------------------------------------------------------------
#define BM 256
#define BN 128
#define BK 32
#define LDSK 40   // padded K stride (bf16 elems)

template<bool B_KMAJOR, bool OUT_BF16>
__global__ __launch_bounds__(256)
void wmma_gemm_kernel(const unsigned short* __restrict__ Ah,
                      const unsigned short* __restrict__ Bh,
                      void* __restrict__ Cp,
                      int Kd, int lda, int ldb, int ldc,
                      long long strideA, long long strideB, long long strideC)
{
    __shared__ unsigned short As[2][BM][LDSK];
    __shared__ unsigned short Bs[2][BN][LDSK];   // B^T layout: [n][k]

    const int tid  = threadIdx.x;
    const int lane = tid & 31;
    const int wave = tid >> 5;         // 0..7
    const int bm   = blockIdx.y * BM;
    const int bn   = blockIdx.x * BN;
    const int batch = blockIdx.z;

    const unsigned short* Ab = Ah + (size_t)batch * (size_t)strideA;  // typed, global AS
    const unsigned short* Bb = Bh + (size_t)batch * (size_t)strideB;  // typed, global AS
    const unsigned long long Abase = (unsigned long long)(uintptr_t)Ab;
    const unsigned long long Bbase = (unsigned long long)(uintptr_t)Bb;
    const size_t cbase = (size_t)batch * (size_t)strideC;

    const int tm   = (wave & 3) * 64;  // wave row offset inside block tile
    const int tn   = (wave >> 2) * 64; // wave col offset inside block tile
    const int lrow = lane & 15;
    const int lhi  = lane >> 4;        // 0 or 1
    const int kbA  = lhi * 8;          // A frag: lanes16-31 start at K=8
    const int kbB  = lhi * 8;          // B frag: lanes16-31 start at K=8

    uint4 gbh[2];  // staging only for the B transpose path

    // ---- async copy of tile k0 into LDS buffer (A always; B when !B_KMAJOR) ----
    auto async_tiles = [&](int k0, int buf) {
        #pragma unroll
        for (int p = 0; p < 4; ++p) {
            const int r = p * 64 + (tid >> 2);
            const int c = (tid & 3) * 8;
            async_b128(lds_off(&As[buf][r][c]),
                       (unsigned)(((size_t)(bm + r) * lda + k0 + c) * 2), Abase);
        }
        if constexpr (!B_KMAJOR) {
            #pragma unroll
            for (int p = 0; p < 2; ++p) {
                const int r = p * 64 + (tid >> 2);
                const int c = (tid & 3) * 8;
                async_b128(lds_off(&Bs[buf][r][c]),
                           (unsigned)(((size_t)(bn + r) * ldb + k0 + c) * 2), Bbase);
            }
        }
    };
    // ---- register-staged B transpose path (typed global pointer) ----
    auto gloadB = [&](int k0) {
        if constexpr (B_KMAJOR) {
            const int kp = tid >> 4;          // 0..15 pair of K-rows
            const int nb = (tid & 15) * 8;    // 0..120
            gbh[0] = *(const uint4*)(Bb + (size_t)(k0 + 2*kp)     * ldb + bn + nb);
            gbh[1] = *(const uint4*)(Bb + (size_t)(k0 + 2*kp + 1) * ldb + bn + nb);
        }
    };
    auto lstoreB = [&](int buf) {
        if constexpr (B_KMAJOR) {
            const int kp = tid >> 4;
            const int nb = (tid & 15) * 8;
            U4U16 u0, u1; u0.q = gbh[0]; u1.q = gbh[1];
            #pragma unroll
            for (int e = 0; e < 8; ++e) {
                const unsigned int pk =
                    (unsigned int)u0.s[e] | ((unsigned int)u1.s[e] << 16);
                *(unsigned int*)&Bs[buf][nb + e][2 * kp] = pk;
            }
        }
    };

    v8f acc[4][4] = {};

    const int nIter = Kd / BK;
    async_tiles(0, 0);
    gloadB(0);
    lstoreB(0);
    wait_async0();
    __syncthreads();

    for (int it = 0; it < nIter; ++it) {
        const int buf = it & 1;
        if (it + 1 < nIter) { async_tiles((it + 1) * BK, 1 - buf); gloadB((it + 1) * BK); }

        // ---- fragments + 16 WMMAs per wave per K-step ----
        Frag16 a[4], b[4];
        #pragma unroll
        for (int i = 0; i < 4; ++i) {
            const int row = tm + i * 16 + lrow;
            a[i].u[0] = *(const uint4*)&As[buf][row][kbA];
            a[i].u[1] = *(const uint4*)&As[buf][row][kbA + 16];
        }
        #pragma unroll
        for (int j = 0; j < 4; ++j) {
            const int nn = tn + j * 16 + lrow;
            b[j].u[0] = *(const uint4*)&Bs[buf][nn][kbB];
            b[j].u[1] = *(const uint4*)&Bs[buf][nn][kbB + 16];
        }
        #pragma unroll
        for (int i = 0; i < 4; ++i)
            #pragma unroll
            for (int j = 0; j < 4; ++j)
                acc[i][j] = __builtin_amdgcn_wmma_f32_16x16x32_bf16(
                    false, a[i].v, false, b[j].v, (short)0, acc[i][j], false, false);

        if (it + 1 < nIter) lstoreB(1 - buf);
        wait_async0();
        __syncthreads();
    }

    // ---- write back (C 16x16 f32 layout: VGPR r -> M=r / M=8+r) ----
    unsigned short* Chs = (unsigned short*)Cp;
    float*          Cfs = (float*)Cp;
    #pragma unroll
    for (int i = 0; i < 4; ++i) {
        const int mb = bm + tm + i * 16 + lhi * 8;
        #pragma unroll
        for (int j = 0; j < 4; ++j) {
            const int n = bn + tn + j * 16 + lrow;
            #pragma unroll
            for (int r = 0; r < 8; ++r) {
                const size_t idx = cbase + (size_t)(mb + r) * ldc + n;
                if constexpr (OUT_BF16) Chs[idx] = f2bfu(acc[i][j][r]);
                else                    Cfs[idx] = acc[i][j][r];
            }
        }
    }
}

// ---------------------------------------------------------------------------
// Row softmax over full LK, then POST-softmax mask (denominator keeps masked
// keys, matching reference), quantize P to bf16. One block per (b, q) row.
// ---------------------------------------------------------------------------
__global__ __launch_bounds__(256)
void softmax_mask_kernel(const float* __restrict__ S, unsigned short* __restrict__ P,
                         const int* __restrict__ VL)
{
    __shared__ float red[256];
    const int row = blockIdx.x;                  // 0 .. B*LQ-1
    const int b   = row / LQ;
    const int vl  = VL[b];
    const float* s = S + (size_t)row * LK;
    unsigned short* p = P + (size_t)row * LK;
    const int tid = threadIdx.x;

    float v[LK / 256];
    float m = -3.4e38f;
    #pragma unroll
    for (int i = 0; i < LK / 256; ++i) {
        v[i] = s[tid + i * 256];
        m = fmaxf(m, v[i]);
    }
    red[tid] = m; __syncthreads();
    for (int off = 128; off > 0; off >>= 1) {
        if (tid < off) red[tid] = fmaxf(red[tid], red[tid + off]);
        __syncthreads();
    }
    m = red[0]; __syncthreads();

    float sum = 0.f;
    #pragma unroll
    for (int i = 0; i < LK / 256; ++i) { v[i] = __expf(v[i] - m); sum += v[i]; }
    red[tid] = sum; __syncthreads();
    for (int off = 128; off > 0; off >>= 1) {
        if (tid < off) red[tid] += red[tid + off];
        __syncthreads();
    }
    const float inv = 1.0f / red[0];

    #pragma unroll
    for (int i = 0; i < LK / 256; ++i) {
        const int col = tid + i * 256;
        const float pv = (col < vl) ? v[i] * inv : 0.0f;
        p[col] = f2bfu(pv);
    }
}

// ---------------------------------------------------------------------------
extern "C" void kernel_launch(void* const* d_in, const int* in_sizes, int n_in,
                              void* d_out, int out_size, void* d_ws, size_t ws_size,
                              hipStream_t stream)
{
    const float* query = (const float*)d_in[0];
    const float* key   = (const float*)d_in[1];
    const float* value = (const float*)d_in[2];
    const int*   vlen  = (const int*)d_in[3];
    const float* Wq    = (const float*)d_in[4];
    const float* Wk    = (const float*)d_in[5];
    const float* Wv    = (const float*)d_in[6];
    float* out = (float*)d_out;

    // Workspace layout (1.13 GiB total):
    //   [0, 402MB)          : Qh/Kh/Vh  (projection outputs, bf16)  live steps 2..5
    //   [402MB, 1170MB)     : union region
    //     phase A (steps 1-2): Xq/Xk/Xv (bf16 inputs) + Wqh/Wkh/Wvh (bf16 weights)
    //     phase B (steps 3-5): S (f32 scores) + P (bf16 probs)
    const size_t nTok = (size_t)BB * LQ * DD;       // 67,108,864
    const size_t nW   = (size_t)DD * DD;            // 1,048,576
    const size_t nS   = (size_t)BB * LQ * LK;       // 134,217,728

    unsigned short* Qh = (unsigned short*)d_ws;
    unsigned short* Kh = Qh + nTok;
    unsigned short* Vh = Kh + nTok;
    unsigned short* U  = Vh + nTok;                 // union region base
    // phase A
    unsigned short* Xq  = U;
    unsigned short* Xk  = Xq + nTok;
    unsigned short* Xv  = Xk + nTok;
    unsigned short* Wqh = Xv + nTok;
    unsigned short* Wkh = Wqh + nW;
    unsigned short* Wvh = Wkh + nW;
    // phase B
    float*          S = (float*)U;
    unsigned short* P = (unsigned short*)(S + nS);

    dim3 blk(256);

    // 1) Convert all f32 operands to bf16 once (streaming, RNE + v_perm pack)
    cvt_f32_bf16_kernel<<<nTok / (8 * 256), blk, 0, stream>>>(query, Xq);
    cvt_f32_bf16_kernel<<<nTok / (8 * 256), blk, 0, stream>>>(key,   Xk);
    cvt_f32_bf16_kernel<<<nTok / (8 * 256), blk, 0, stream>>>(value, Xv);
    cvt_f32_bf16_kernel<<<nW   / (8 * 256), blk, 0, stream>>>(Wq, Wqh);
    cvt_f32_bf16_kernel<<<nW   / (8 * 256), blk, 0, stream>>>(Wk, Wkh);
    cvt_f32_bf16_kernel<<<nW   / (8 * 256), blk, 0, stream>>>(Wv, Wvh);

    // 2) Projections: [B*LQ, D] x [D, D] -> bf16
    dim3 gp(DD / BN, (BB * LQ) / BM, 1);            // (8, 256)
    wmma_gemm_kernel<true, true><<<gp, blk, 0, stream>>>(
        Xq, Wqh, Qh, DD, DD, DD, DD, 0, 0, 0);
    wmma_gemm_kernel<true, true><<<gp, blk, 0, stream>>>(
        Xk, Wkh, Kh, DD, DD, DD, DD, 0, 0, 0);
    wmma_gemm_kernel<true, true><<<gp, blk, 0, stream>>>(
        Xv, Wvh, Vh, DD, DD, DD, DD, 0, 0, 0);

    // 3) Scores: S[b] = Q[b] * K[b]^T   (both bf16 [L][D], out f32)
    dim3 gs(LK / BN, LQ / BM, BB);                  // (16, 8, 32)
    wmma_gemm_kernel<false, false><<<gs, blk, 0, stream>>>(
        Qh, Kh, S, DD, DD, DD, LK,
        (long long)LQ * DD, (long long)LK * DD, (long long)LQ * LK);

    // 4) Softmax over LK + post-softmax mask + bf16 quantize
    softmax_mask_kernel<<<BB * LQ, 256, 0, stream>>>(S, P, vlen);

    // 5) Out: out[b] = P[b] * V[b]   (P bf16 [LQ][LK], V bf16 [LK][D] K-major, out f32)
    dim3 gv(DD / BN, LQ / BM, BB);                  // (8, 8, 32)
    wmma_gemm_kernel<true, false><<<gv, blk, 0, stream>>>(
        P, Vh, out, LK, LK, DD, DD,
        (long long)LQ * LK, (long long)LK * DD, (long long)LQ * DD);
}
